// Attention_32186484917066
// MI455X (gfx1250) — matrix operand order repaired and checked by reading, not compiled, not run on hardware
//
#include <hip/hip_runtime.h>
#include <hip/hip_bf16.h>
#include <stdint.h>

// ---------------------------------------------------------------------------
// CDNA5 (gfx1250) causal attention layer
//   wave32 + v_wmma_f32_16x16x32_bf16 + global_load_async_to_lds_b128
// ---------------------------------------------------------------------------

typedef __attribute__((ext_vector_type(16))) __bf16 v16bf;
typedef __attribute__((ext_vector_type(8)))  float  v8f;

#define WMMA_BF16(A, B, C) \
    __builtin_amdgcn_wmma_f32_16x16x32_bf16(false, (A), false, (B), (short)0, (C), false, false)

static constexpr int S  = 4096;
static constexpr int D  = 1024;
static constexpr int H  = 16;
static constexpr int DH = 64;

// ---------------------------------------------------------------------------
// Async global->LDS copy (gfx1250 VGLOBAL async path, tracked by ASYNCcnt).
// GV form: vdst = LDS byte address (low 32 bits of generic LDS pointer),
//          vaddr = 64-bit global address VGPR pair.
// ---------------------------------------------------------------------------
__device__ inline void async_copy_b128(__bf16* lds, const __bf16* g) {
    const uint32_t laddr = (uint32_t)(uintptr_t)lds;
    asm volatile("global_load_async_to_lds_b128 %0, %1, off"
                 :: "v"(laddr), "v"(g) : "memory");
}

__device__ inline void wait_async0() {
    asm volatile("s_wait_asynccnt 0x0" ::: "memory");
}

// ---------------------------------------------------------------------------
// Fragment loaders (layouts per CDNA5 ISA 7.12.2, wave32)
// ---------------------------------------------------------------------------
__device__ inline v16bf load_frag_a(const __bf16* base, int ld, int lane) {
    const int m  = lane & 15;
    const int kh = (lane >> 4) & 1;
    const __bf16* row = base + m * ld;
    v16bf a;
    uint32_t* ap = (uint32_t*)&a;
#pragma unroll
    for (int i = 0; i < 4; ++i) {
        ap[i]     = *(const uint32_t*)(row + kh * 8 + 2 * i);
        ap[i + 4] = *(const uint32_t*)(row + 16 + kh * 8 + 2 * i);
    }
    return a;
}

// B 32x16 bf16, element (k, n) at base[n*ld + k] (K contiguous).
__device__ inline v16bf load_frag_b(const __bf16* base, int ld, int lane) {
    const int n  = lane & 15;
    const int kh = (lane >> 4) & 1;
    const __bf16* col = base + n * ld;
    v16bf b;
    uint32_t* bp = (uint32_t*)&b;
#pragma unroll
    for (int i = 0; i < 4; ++i) {
        bp[i]     = *(const uint32_t*)(col + kh * 8 + 2 * i);
        bp[i + 4] = *(const uint32_t*)(col + 16 + kh * 8 + 2 * i);
    }
    return b;
}

// ---------------------------------------------------------------------------
// Kernel 1: QKV projection with register-staged double buffering.
//   Qb/Kb: [H][S][DH] bf16 (Q pre-scaled by 1/8); Vb TRANSPOSED: [H][DH][S]
// ---------------------------------------------------------------------------
__global__ __launch_bounds__(256) void qkv_gemm_kernel(
    const float* __restrict__ x, const float* __restrict__ W,
    const float* __restrict__ bias,
    __bf16* __restrict__ Qb, __bf16* __restrict__ Kb, __bf16* __restrict__ Vb)
{
    constexpr int BM = 128, BN = 128, BK = 32;
    constexpr int LA = BK + 8, LB = BK + 8;       // even pads
    __shared__ __bf16 sA[BM * LA];
    __shared__ __bf16 sB[BN * LB];                // [n][k]

    const int tid  = threadIdx.x;
    const int lane = tid & 31;
    const int wave = tid >> 5;
    const int wm   = wave & 3;
    const int wn   = wave >> 2;
    const int row0 = blockIdx.y * BM;
    const int col0 = blockIdx.x * BN;

    v8f acc[2][4];
#pragma unroll
    for (int i = 0; i < 2; ++i)
#pragma unroll
        for (int j = 0; j < 4; ++j) acc[i][j] = v8f{};

    float4 ra[4], rb[4];
    // prologue: stage k0 = 0
#pragma unroll
    for (int i = 0; i < 4; ++i) {
        const int lin = (tid + i * 256) * 4;
        ra[i] = *(const float4*)(x + (size_t)(row0 + (lin >> 5)) * D + (lin & 31));
        rb[i] = *(const float4*)(W + (size_t)(lin >> 7) * (3 * D) + col0 + (lin & 127));
    }

    for (int k0 = 0; k0 < D; k0 += BK) {
        // commit staged registers to LDS (fp32 -> bf16)
#pragma unroll
        for (int i = 0; i < 4; ++i) {
            const int lin = (tid + i * 256) * 4;
            __bf16* da = sA + (lin >> 5) * LA + (lin & 31);
            da[0] = (__bf16)ra[i].x; da[1] = (__bf16)ra[i].y;
            da[2] = (__bf16)ra[i].z; da[3] = (__bf16)ra[i].w;
            const int kr = lin >> 7, nc = lin & 127;
            sB[(nc + 0) * LB + kr] = (__bf16)rb[i].x;
            sB[(nc + 1) * LB + kr] = (__bf16)rb[i].y;
            sB[(nc + 2) * LB + kr] = (__bf16)rb[i].z;
            sB[(nc + 3) * LB + kr] = (__bf16)rb[i].w;
        }
        __syncthreads();

        // prefetch next tile while this tile's WMMAs run
        const int kn = k0 + BK;
        if (kn < D) {
#pragma unroll
            for (int i = 0; i < 4; ++i) {
                const int lin = (tid + i * 256) * 4;
                ra[i] = *(const float4*)(x + (size_t)(row0 + (lin >> 5)) * D + kn + (lin & 31));
                rb[i] = *(const float4*)(W + (size_t)(kn + (lin >> 7)) * (3 * D) + col0 + (lin & 127));
            }
        }

        const v16bf a0 = load_frag_a(sA + (wm * 32 + 0)  * LA, LA, lane);
        const v16bf a1 = load_frag_a(sA + (wm * 32 + 16) * LA, LA, lane);
        v16bf bfr[4];
#pragma unroll
        for (int t = 0; t < 4; ++t)
            bfr[t] = load_frag_b(sB + (wn * 64 + t * 16) * LB, LB, lane);
#pragma unroll
        for (int t = 0; t < 4; ++t) {
            acc[0][t] = WMMA_BF16(a0, bfr[t], acc[0][t]);
            acc[1][t] = WMMA_BF16(a1, bfr[t], acc[1][t]);
        }
        __syncthreads();
    }

    const float qscale = 0.125f;   // 1/sqrt(64)
    const int n_lane = lane & 15;
    const int mh     = lane >> 4;
#pragma unroll
    for (int mt = 0; mt < 2; ++mt) {
#pragma unroll
        for (int nt = 0; nt < 4; ++nt) {
            const int ncol = col0 + wn * 64 + nt * 16 + n_lane;
            const float bv = bias[ncol];
#pragma unroll
            for (int r = 0; r < 8; ++r) {
                const int srow = row0 + wm * 32 + mt * 16 + r + 8 * mh;
                float v = acc[mt][nt][r] + bv;
                const int n = ncol;
                if (n < D) {                       // Q: scaled, head-major
                    Qb[(size_t)(n >> 6) * S * DH + (size_t)srow * DH + (n & 63)] =
                        (__bf16)(v * qscale);
                } else if (n < 2 * D) {            // K: head-major row-major
                    const int nk = n - D;
                    Kb[(size_t)(nk >> 6) * S * DH + (size_t)srow * DH + (nk & 63)] =
                        (__bf16)v;
                } else {                           // V: head-major TRANSPOSED [H][DH][S]
                    const int nv = n - 2 * D;
                    Vb[(size_t)(nv >> 6) * DH * S + (size_t)(nv & 63) * S + srow] =
                        (__bf16)v;
                }
            }
        }
    }
}

// ---------------------------------------------------------------------------
// Kernel 2: causal flash attention. One block per (q-block of 64, head),
// 4 waves, double-buffered async K/Vt staging.
// ---------------------------------------------------------------------------
static constexpr int LK = DH + 8;   // 72 elements = 144 B row stride (16B mult)

__device__ inline void stage_kv_async(const __bf16* kp, const __bf16* vtp,
                                      __bf16* sKb, __bf16* sVb, int tid) {
    // K block: 64 rows x 64 bf16 (128B) = 512 x 16B chunks; same for Vt.
#pragma unroll
    for (int i = 0; i < 4; ++i) {
        const int idx = tid + i * 128;           // 0..511
        const int rr = idx >> 3;
        const int ce = (idx & 7) * 8;            // element offset within row
        async_copy_b128(sKb + rr * LK + ce, kp + (size_t)rr * DH + ce);
        async_copy_b128(sVb + rr * LK + ce, vtp + (size_t)rr * S + ce);
    }
}

__global__ __launch_bounds__(128) void flash_attn_kernel(
    const __bf16* __restrict__ Qb, const __bf16* __restrict__ Kb,
    const __bf16* __restrict__ Vb, __bf16* __restrict__ Ob)
{
    constexpr int BQ = 64, BC = 64;
    __shared__ __bf16 sK [2][BC * LK];          // (kv, d) row-major
    __shared__ __bf16 sVt[2][DH * LK];          // (d, kv) K-contiguous
    __shared__ __bf16 sP [4 * 16 * LK];         // per-wave P scratch

    const int tid  = threadIdx.x;
    const int lane = tid & 31;
    const int wave = tid >> 5;
    const int qb   = blockIdx.x;
    const int h    = blockIdx.y;
    const int n_lane = lane & 15;
    const int kh     = (lane >> 4) & 1;         // also mh for C layout

    const __bf16* Qh  = Qb + (size_t)h * S * DH;
    const __bf16* Kh  = Kb + (size_t)h * S * DH;
    const __bf16* Vth = Vb + (size_t)h * DH * S;    // transposed [DH][S]

    // Q fragments straight from global
    v16bf aq[2];
    {
        const int qrow = qb * BQ + wave * 16 + n_lane;
        const __bf16* qp = Qh + (size_t)qrow * DH;
#pragma unroll
        for (int f = 0; f < 2; ++f) {
            uint32_t* ap = (uint32_t*)&aq[f];
            const int kb = f * 32;
#pragma unroll
            for (int i = 0; i < 4; ++i) {
                ap[i]     = *(const uint32_t*)(qp + kb + kh * 8 + 2 * i);
                ap[i + 4] = *(const uint32_t*)(qp + kb + 16 + kh * 8 + 2 * i);
            }
        }
    }

    v8f o[4];
#pragma unroll
    for (int t = 0; t < 4; ++t) o[t] = v8f{};
    float mrow[8], lrow[8];
#pragma unroll
    for (int r = 0; r < 8; ++r) { mrow[r] = -3.0e38f; lrow[r] = 0.0f; }

    // prefetch block 0
    stage_kv_async(Kh, Vth, sK[0], sVt[0], tid);

    for (int j = 0; j <= qb; ++j) {
        const int cur = j & 1;
        wait_async0();          // my async chunks have landed in LDS
        __syncthreads();        // everyone's chunks have landed

        if (j < qb)             // overlap next block's DMA with this block's math
            stage_kv_async(Kh + (size_t)(j + 1) * BC * DH, Vth + (size_t)(j + 1) * BC,
                           sK[cur ^ 1], sVt[cur ^ 1], tid);

        // S = Q @ K^T
        v8f sc[4];
#pragma unroll
        for (int t = 0; t < 4; ++t) {
            v8f s = v8f{};
            const v16bf bk0 = load_frag_b(sK[cur] + (t * 16) * LK,      LK, lane);
            const v16bf bk1 = load_frag_b(sK[cur] + (t * 16) * LK + 32, LK, lane);
            s = WMMA_BF16(aq[0], bk0, s);
            s = WMMA_BF16(aq[1], bk1, s);
            sc[t] = s;
        }

        // causal mask on diagonal block
        if (j == qb) {
#pragma unroll
            for (int t = 0; t < 4; ++t)
#pragma unroll
                for (int r = 0; r < 8; ++r) {
                    const int gq = qb * 64 + wave * 16 + r + 8 * kh;
                    const int gk = j * 64 + t * 16 + n_lane;
                    if (gk > gq) sc[t][r] = -1.0e30f;
                }
        }

        // online softmax (each 16-lane half owns rows r+8*kh)
        float mnew[8], alpha[8];
#pragma unroll
        for (int r = 0; r < 8; ++r) {
            float m = mrow[r];
#pragma unroll
            for (int t = 0; t < 4; ++t) m = fmaxf(m, sc[t][r]);
#pragma unroll
            for (int dlt = 1; dlt < 16; dlt <<= 1)
                m = fmaxf(m, __shfl_xor(m, dlt, 32));
            mnew[r]  = m;
            alpha[r] = __expf(mrow[r] - m);
        }
#pragma unroll
        for (int r = 0; r < 8; ++r) {
            float rs = 0.0f;
#pragma unroll
            for (int t = 0; t < 4; ++t) {
                const float p = __expf(sc[t][r] - mnew[r]);
                sc[t][r] = p;
                rs += p;
            }
#pragma unroll
            for (int dlt = 1; dlt < 16; dlt <<= 1)
                rs += __shfl_xor(rs, dlt, 32);
            lrow[r] = lrow[r] * alpha[r] + rs;
            mrow[r] = mnew[r];
        }
#pragma unroll
        for (int t = 0; t < 4; ++t)
#pragma unroll
            for (int r = 0; r < 8; ++r) o[t][r] *= alpha[r];

        // spill P to per-wave LDS scratch, reload as A frags (same-wave DScnt dep)
        __bf16* pw = sP + wave * 16 * LK;
#pragma unroll
        for (int t = 0; t < 4; ++t)
#pragma unroll
            for (int r = 0; r < 8; ++r)
                pw[(r + 8 * kh) * LK + t * 16 + n_lane] = (__bf16)sc[t][r];

        const v16bf ap0 = load_frag_a(pw,      LK, lane);
        const v16bf ap1 = load_frag_a(pw + 32, LK, lane);

        // O += P @ V
#pragma unroll
        for (int t = 0; t < 4; ++t) {
            const v16bf bv0 = load_frag_b(sVt[cur] + (t * 16) * LK,      LK, lane);
            const v16bf bv1 = load_frag_b(sVt[cur] + (t * 16) * LK + 32, LK, lane);
            o[t] = WMMA_BF16(ap0, bv0, o[t]);
            o[t] = WMMA_BF16(ap1, bv1, o[t]);
        }
    }

    // normalize and store merged-head bf16 [S][D]
#pragma unroll
    for (int r = 0; r < 8; ++r) {
        const float inv = 1.0f / lrow[r];
        const int srow = qb * 64 + wave * 16 + r + 8 * kh;
#pragma unroll
        for (int t = 0; t < 4; ++t)
            Ob[(size_t)srow * D + h * DH + t * 16 + n_lane] = (__bf16)(o[t][r] * inv);
    }
}

// ---------------------------------------------------------------------------
// Kernel 3: output projection (register-staged double buffering).
// ---------------------------------------------------------------------------
__global__ __launch_bounds__(256) void proj_gemm_kernel(
    const __bf16* __restrict__ A, const float* __restrict__ W,
    const float* __restrict__ bias, float* __restrict__ out)
{
    constexpr int BM = 128, BN = 128, BK = 32;
    constexpr int LA = BK + 8, LB = BK + 8;
    __shared__ __bf16 sA[BM * LA];
    __shared__ __bf16 sB[BN * LB];

    const int tid  = threadIdx.x;
    const int lane = tid & 31;
    const int wave = tid >> 5;
    const int wm   = wave & 3;
    const int wn   = wave >> 2;
    const int row0 = blockIdx.y * BM;
    const int col0 = blockIdx.x * BN;

    v8f acc[2][4];
#pragma unroll
    for (int i = 0; i < 2; ++i)
#pragma unroll
        for (int j = 0; j < 4; ++j) acc[i][j] = v8f{};

    uint32_t rau[8];
    float4   rb[4];
#pragma unroll
    for (int i = 0; i < 2; ++i) {
        const int lin = (tid + i * 256) * 8;
        const uint32_t* src = (const uint32_t*)(A + (size_t)(row0 + (lin >> 5)) * D + (lin & 31));
#pragma unroll
        for (int q = 0; q < 4; ++q) rau[i * 4 + q] = src[q];
    }
#pragma unroll
    for (int i = 0; i < 4; ++i) {
        const int lin = (tid + i * 256) * 4;
        rb[i] = *(const float4*)(W + (size_t)(lin >> 7) * D + col0 + (lin & 127));
    }

    for (int k0 = 0; k0 < D; k0 += BK) {
#pragma unroll
        for (int i = 0; i < 2; ++i) {
            const int lin = (tid + i * 256) * 8;
            uint32_t* dst = (uint32_t*)(sA + (lin >> 5) * LA + (lin & 31));
#pragma unroll
            for (int q = 0; q < 4; ++q) dst[q] = rau[i * 4 + q];
        }
#pragma unroll
        for (int i = 0; i < 4; ++i) {
            const int lin = (tid + i * 256) * 4;
            const int kr = lin >> 7, nc = lin & 127;
            sB[(nc + 0) * LB + kr] = (__bf16)rb[i].x;
            sB[(nc + 1) * LB + kr] = (__bf16)rb[i].y;
            sB[(nc + 2) * LB + kr] = (__bf16)rb[i].z;
            sB[(nc + 3) * LB + kr] = (__bf16)rb[i].w;
        }
        __syncthreads();

        const int kn = k0 + BK;
        if (kn < D) {
#pragma unroll
            for (int i = 0; i < 2; ++i) {
                const int lin = (tid + i * 256) * 8;
                const uint32_t* src =
                    (const uint32_t*)(A + (size_t)(row0 + (lin >> 5)) * D + kn + (lin & 31));
#pragma unroll
                for (int q = 0; q < 4; ++q) rau[i * 4 + q] = src[q];
            }
#pragma unroll
            for (int i = 0; i < 4; ++i) {
                const int lin = (tid + i * 256) * 4;
                rb[i] = *(const float4*)(W + (size_t)(kn + (lin >> 7)) * D + col0 + (lin & 127));
            }
        }

        const v16bf a0 = load_frag_a(sA + (wm * 32 + 0)  * LA, LA, lane);
        const v16bf a1 = load_frag_a(sA + (wm * 32 + 16) * LA, LA, lane);
        v16bf bfr[4];
#pragma unroll
        for (int t = 0; t < 4; ++t)
            bfr[t] = load_frag_b(sB + (wn * 64 + t * 16) * LB, LB, lane);
#pragma unroll
        for (int t = 0; t < 4; ++t) {
            acc[0][t] = WMMA_BF16(a0, bfr[t], acc[0][t]);
            acc[1][t] = WMMA_BF16(a1, bfr[t], acc[1][t]);
        }
        __syncthreads();
    }

    const int n_lane = lane & 15;
    const int mh     = lane >> 4;
#pragma unroll
    for (int mt = 0; mt < 2; ++mt) {
#pragma unroll
        for (int nt = 0; nt < 4; ++nt) {
            const int ncol = col0 + wn * 64 + nt * 16 + n_lane;
            const float bv = bias[ncol];
#pragma unroll
            for (int r = 0; r < 8; ++r) {
                const int srow = row0 + wm * 32 + mt * 16 + r + 8 * mh;
                out[(size_t)srow * D + ncol] = acc[mt][nt][r] + bv;
            }
        }
    }
}

// ---------------------------------------------------------------------------
// Launch. Workspace (bf16): [0,8M) Q [H][S][64]; [8M,16M) K [H][S][64];
//   [16M,24M) V transposed [H][64][S]; [24M,32M) attn out [S][D]
// ---------------------------------------------------------------------------
extern "C" void kernel_launch(void* const* d_in, const int* in_sizes, int n_in,
                              void* d_out, int out_size, void* d_ws, size_t ws_size,
                              hipStream_t stream) {
    const float* x     = (const float*)d_in[0];
    const float* Wqkv  = (const float*)d_in[1];
    const float* bqkv  = (const float*)d_in[2];
    const float* Wproj = (const float*)d_in[3];
    const float* bproj = (const float*)d_in[4];
    float* out = (float*)d_out;

    char* ws = (char*)d_ws;
    const size_t MB8 = (size_t)8 * 1024 * 1024;
    __bf16* Qb = (__bf16*)(ws + 0 * MB8);
    __bf16* Kb = (__bf16*)(ws + 1 * MB8);
    __bf16* Vb = (__bf16*)(ws + 2 * MB8);
    __bf16* Ob = (__bf16*)(ws + 3 * MB8);

    dim3 g1((3 * D) / 128, S / 128);
    qkv_gemm_kernel<<<g1, 256, 0, stream>>>(x, Wqkv, bqkv, Qb, Kb, Vb);

    dim3 g2(S / 64, H);
    flash_attn_kernel<<<g2, 128, 0, stream>>>(Qb, Kb, Vb, Ob);

    dim3 g3(D / 128, S / 128);
    proj_gemm_kernel<<<g3, 256, 0, stream>>>(Ob, Wproj, bproj, out);
}
